// EnhancedGAT_24550033064673
// MI455X (gfx1250) — matrix operand order repaired and checked by reading, not compiled, not run on hardware
//
#include <hip/hip_runtime.h>
#include <hip/hip_bf16.h>

#define NNODES 50000
#define NEDGES 800000
#define EP     850000      // edges + self loops
#define INCH   128
#define HIDC   32
#define OUTC   40
#define H1CH   128         // 4 heads * 32
#define NEG    0.2f
#define KDIM   128         // K for both transform GEMMs
#define MTILES 3125        // 50000 / 16 exactly
#define MTPW   8           // 16-row tiles per wave

typedef __attribute__((ext_vector_type(16))) __bf16 v16bf;
typedef __attribute__((ext_vector_type(8)))  float  v8f;

// ---------- helpers ----------
__device__ __forceinline__ unsigned fenc(float f) {
  unsigned u = __float_as_uint(f);
  return (u & 0x80000000u) ? ~u : (u | 0x80000000u);   // monotone float->uint
}
__device__ __forceinline__ float fdec(unsigned u) {
  unsigned b = (u & 0x80000000u) ? (u & 0x7fffffffu) : ~u;
  return __uint_as_float(b);
}
__device__ __forceinline__ void edge_sd(const int* __restrict__ ei, int e, int& s, int& d) {
  if (e < NEDGES) { s = ei[e]; d = ei[NEDGES + e]; }
  else            { s = e - NEDGES; d = s; }            // self loops appended
}
__device__ __forceinline__ float eluf(float v) {
  return v > 0.f ? v : (__expf(v) - 1.f);
}

// ---------- init ----------
__global__ void zero_f32(float* __restrict__ p, long n) {
  long i = (long)blockIdx.x * blockDim.x + threadIdx.x;
  long st = (long)gridDim.x * blockDim.x;
  for (; i < n; i += st) p[i] = 0.f;
}

// ---------- WMMA A-fragment loader (fp32 row-major -> bf16 regs) ----------
__device__ __forceinline__ v16bf load_a_bf16(const float* __restrict__ A, int lda,
                                             int row0, int k0) {
  int lane = threadIdx.x & 31;
  int m  = lane & 15;
  int kb = (lane >> 4) << 3;                 // 0 or 8
  const float* p = A + (size_t)(row0 + m) * lda + k0 + kb;
  v16bf a;
#pragma unroll
  for (int i = 0; i < 8; ++i) a[i] = (__bf16)p[i];
#pragma unroll
  for (int i = 0; i < 8; ++i) a[i + 8] = (__bf16)p[i + 16];
  return a;
}

// ---------- dual GEMM: out{L,R} = A @ W{l,r} + b{l,r}; K = 128 ----------
// B fragments preloaded once per wave and reused over MTPW row tiles.
__global__ __launch_bounds__(32)
void gemm_dual(const float* __restrict__ A,
               const float* __restrict__ Wl, const float* __restrict__ bl, float* __restrict__ outL,
               const float* __restrict__ Wr, const float* __restrict__ br, float* __restrict__ outR,
               int NC) {
  int ct = blockIdx.y;
  int ntPer = NC >> 4;
  int mat = ct / ntPer;
  int n0  = (ct % ntPer) << 4;
  const float* W    = mat ? Wr : Wl;
  const float* bias = mat ? br : bl;
  float*       out  = mat ? outR : outL;

  int lane = threadIdx.x & 31;
  int n  = lane & 15;
  int kb = (lane >> 4) << 3;                 // 0 or 8

  // preload all 4 B fragments (unconditional strided loads, no branches)
  v16bf bf[4];
  {
    const float* wp = W + (size_t)kb * NC + n0 + n;
#pragma unroll
    for (int kt = 0; kt < 4; ++kt) {
#pragma unroll
      for (int i = 0; i < 16; ++i)
        bf[kt][i] = (__bf16)wp[(size_t)(kt * 32 + i + (i >= 8 ? 8 : 0)) * NC];
    }
  }
  float bv  = bias[n0 + n];
  int rbias = (lane >> 4) << 3;              // row offset 0 or 8 for D

  int mt    = blockIdx.x * MTPW;
  int mtEnd = mt + MTPW;
  if (mtEnd > MTILES) mtEnd = MTILES;
  for (; mt < mtEnd; ++mt) {
    int row0 = mt << 4;
    v8f c = {};
#pragma unroll
    for (int kt = 0; kt < 4; ++kt) {
      v16bf a = load_a_bf16(A, KDIM, row0, kt * 32);
      c = __builtin_amdgcn_wmma_f32_16x16x32_bf16(false, a, false, bf[kt], (short)0, c, false, false);
    }
    int col   = n0 + n;
    int rbase = row0 + rbias;
#pragma unroll
    for (int r = 0; r < 8; ++r)
      out[(size_t)(rbase + r) * NC + col] = c[r] + bv;
  }
}

// ---------- classifier GEMM: out = [h2 | x] @ Wc + bc; K = 160 ----------
__global__ __launch_bounds__(32)
void gemm_cls(const float* __restrict__ H2, const float* __restrict__ X,
              const float* __restrict__ Wc, const float* __restrict__ bc,
              float* __restrict__ out) {
  int n0   = blockIdx.y << 4;
  int lane = threadIdx.x & 31;
  int n    = lane & 15;
  int col  = n0 + n;
  bool ok  = col < OUTC;
  int colc = ok ? col : (OUTC - 1);          // clamped, always-valid index

  // preload 5 B fragments; unconditional loads from clamped col, zero via select
  v16bf bf[5];
  {
    int kb = (lane >> 4) << 3;               // 0 or 8
    const float* wp = Wc + (size_t)kb * OUTC + colc;
#pragma unroll
    for (int kt = 0; kt < 5; ++kt) {
#pragma unroll
      for (int i = 0; i < 16; ++i) {
        float v = wp[(size_t)(kt * 32 + i + (i >= 8 ? 8 : 0)) * OUTC];
        bf[kt][i] = (__bf16)(ok ? v : 0.f);
      }
    }
  }
  float bv  = bc[colc];
  int m     = lane & 15;
  int kba   = (lane >> 4) << 3;              // 0 or 8
  int rbias = (lane >> 4) << 3;

  int mt    = blockIdx.x * MTPW;
  int mtEnd = mt + MTPW;
  if (mtEnd > MTILES) mtEnd = MTILES;
  for (; mt < mtEnd; ++mt) {
    int row0 = mt << 4;
    v8f c = {};
#pragma unroll
    for (int kt = 0; kt < 5; ++kt) {
      int k0 = kt * 32;
      const float* p = (kt == 0) ? (H2 + (size_t)(row0 + m) * HIDC + kba)
                                 : (X  + (size_t)(row0 + m) * INCH + (k0 - HIDC) + kba);
      v16bf a;
#pragma unroll
      for (int i = 0; i < 8; ++i) a[i] = (__bf16)p[i];
#pragma unroll
      for (int i = 0; i < 8; ++i) a[i + 8] = (__bf16)p[i + 16];
      c = __builtin_amdgcn_wmma_f32_16x16x32_bf16(false, a, false, bf[kt], (short)0, c, false, false);
    }
    if (ok) {
      int rbase = row0 + rbias;
#pragma unroll
      for (int r = 0; r < 8; ++r)
        out[(size_t)(rbase + r) * OUTC + col] = c[r] + bv;
    }
  }
}

// ---------- conv1 edge passes (heads=4, ch=32) ----------
__global__ void edge1_attn(const int* __restrict__ ei,
                           const float* __restrict__ xl, const float* __restrict__ xr,
                           const float* __restrict__ att,
                           float* __restrict__ elog, unsigned* __restrict__ mbits) {
  int idx = blockIdx.x * blockDim.x + threadIdx.x;
  if (idx >= EP * 4) return;
  int e = idx >> 2, h = idx & 3;
  int s, d; edge_sd(ei, e, s, d);
  const float4* a = (const float4*)(xl + (size_t)s * H1CH + h * HIDC);
  const float4* b = (const float4*)(xr + (size_t)d * H1CH + h * HIDC);
  const float4* w = (const float4*)(att + h * HIDC);
  float acc = 0.f;
#pragma unroll
  for (int j = 0; j < 8; ++j) {
    float4 av = a[j], bv = b[j], wv = w[j];
    float sx = av.x + bv.x; sx = sx > 0.f ? sx : NEG * sx;
    float sy = av.y + bv.y; sy = sy > 0.f ? sy : NEG * sy;
    float sz = av.z + bv.z; sz = sz > 0.f ? sz : NEG * sz;
    float sw = av.w + bv.w; sw = sw > 0.f ? sw : NEG * sw;
    acc += sx * wv.x + sy * wv.y + sz * wv.z + sw * wv.w;
  }
  elog[idx] = acc;
  atomicMax(mbits + (size_t)d * 4 + h, fenc(acc));
}

__global__ void edge1_exp(const int* __restrict__ ei, float* __restrict__ elog,
                          const unsigned* __restrict__ mbits, float* __restrict__ denom) {
  int idx = blockIdx.x * blockDim.x + threadIdx.x;
  if (idx >= EP * 4) return;
  int e = idx >> 2, h = idx & 3;
  int s, d; edge_sd(ei, e, s, d);
  float m  = fdec(mbits[(size_t)d * 4 + h]);
  float ex = __expf(elog[idx] - m);
  elog[idx] = ex;
  atomicAdd(denom + (size_t)d * 4 + h, ex);
}

__global__ void edge1_agg(const int* __restrict__ ei, const float* __restrict__ elog,
                          const float* __restrict__ denom, const float* __restrict__ xl,
                          float* __restrict__ h1raw) {
  int idx = blockIdx.x * blockDim.x + threadIdx.x;
  if (idx >= EP * 32) return;
  int e = idx >> 5, l = idx & 31, h = l >> 3;
  int s, d; edge_sd(ei, e, s, d);
  float alpha = elog[(size_t)e * 4 + h] / (denom[(size_t)d * 4 + h] + 1e-16f);
  float4 v = ((const float4*)(xl + (size_t)s * H1CH))[l];
  float* o = h1raw + (size_t)d * H1CH + l * 4;
  atomicAdd(o + 0, v.x * alpha);
  atomicAdd(o + 1, v.y * alpha);
  atomicAdd(o + 2, v.z * alpha);
  atomicAdd(o + 3, v.w * alpha);
}

// ---------- batchnorm (bias1 cancels under BN) ----------
__global__ void bn_stats(const float* __restrict__ h, float* __restrict__ sum,
                         float* __restrict__ sumsq) {
  int ch = threadIdx.x;                       // 128 channels
  int r0 = blockIdx.x * 128;
  float s = 0.f, q = 0.f;
  int rend = min(r0 + 128, NNODES);
  for (int r = r0; r < rend; ++r) {
    float v = h[(size_t)r * H1CH + ch];
    s += v; q += v * v;
  }
  atomicAdd(sum + ch, s);
  atomicAdd(sumsq + ch, q);
}

__global__ void bn_final(const float* __restrict__ sum, const float* __restrict__ sumsq,
                         const float* __restrict__ gamma, const float* __restrict__ beta,
                         float* __restrict__ ss) {
  int c = threadIdx.x;
  float mu  = sum[c]   * (1.f / NNODES);
  float var = sumsq[c] * (1.f / NNODES) - mu * mu;
  float sc  = gamma[c] * rsqrtf(var + 1e-5f);
  ss[c]       = sc;
  ss[128 + c] = beta[c] - mu * sc;
}

__global__ void bn_apply_elu(float* __restrict__ h, const float* __restrict__ ss) {
  int idx = blockIdx.x * blockDim.x + threadIdx.x;
  if (idx >= NNODES * H1CH) return;
  int ch = idx & (H1CH - 1);
  h[idx] = eluf(h[idx] * ss[ch] + ss[128 + ch]);
}

// ---------- conv2 edge passes (heads=1, ch=32) ----------
__global__ void edge2_attn(const int* __restrict__ ei,
                           const float* __restrict__ xl, const float* __restrict__ xr,
                           const float* __restrict__ att,
                           float* __restrict__ elog, unsigned* __restrict__ mbits) {
  int e = blockIdx.x * blockDim.x + threadIdx.x;
  if (e >= EP) return;
  int s, d; edge_sd(ei, e, s, d);
  const float4* a = (const float4*)(xl + (size_t)s * HIDC);
  const float4* b = (const float4*)(xr + (size_t)d * HIDC);
  const float4* w = (const float4*)att;
  float acc = 0.f;
#pragma unroll
  for (int j = 0; j < 8; ++j) {
    float4 av = a[j], bv = b[j], wv = w[j];
    float sx = av.x + bv.x; sx = sx > 0.f ? sx : NEG * sx;
    float sy = av.y + bv.y; sy = sy > 0.f ? sy : NEG * sy;
    float sz = av.z + bv.z; sz = sz > 0.f ? sz : NEG * sz;
    float sw = av.w + bv.w; sw = sw > 0.f ? sw : NEG * sw;
    acc += sx * wv.x + sy * wv.y + sz * wv.z + sw * wv.w;
  }
  elog[e] = acc;
  atomicMax(mbits + d, fenc(acc));
}

__global__ void edge2_exp(const int* __restrict__ ei, float* __restrict__ elog,
                          const unsigned* __restrict__ mbits, float* __restrict__ denom) {
  int e = blockIdx.x * blockDim.x + threadIdx.x;
  if (e >= EP) return;
  int s, d; edge_sd(ei, e, s, d);
  float ex = __expf(elog[e] - fdec(mbits[d]));
  elog[e] = ex;
  atomicAdd(denom + d, ex);
}

__global__ void edge2_agg(const int* __restrict__ ei, const float* __restrict__ elog,
                          const float* __restrict__ denom, const float* __restrict__ xl,
                          float* __restrict__ h2raw) {
  int idx = blockIdx.x * blockDim.x + threadIdx.x;
  if (idx >= EP * 8) return;
  int e = idx >> 3, l = idx & 7;
  int s, d; edge_sd(ei, e, s, d);
  float alpha = elog[e] / (denom[d] + 1e-16f);
  float4 v = ((const float4*)(xl + (size_t)s * HIDC))[l];
  float* o = h2raw + (size_t)d * HIDC + l * 4;
  atomicAdd(o + 0, v.x * alpha);
  atomicAdd(o + 1, v.y * alpha);
  atomicAdd(o + 2, v.z * alpha);
  atomicAdd(o + 3, v.w * alpha);
}

__global__ void elu2_bias(float* __restrict__ h, const float* __restrict__ bias2) {
  int idx = blockIdx.x * blockDim.x + threadIdx.x;
  if (idx >= NNODES * HIDC) return;
  h[idx] = eluf(h[idx] + bias2[idx & (HIDC - 1)]);
}

// ---------- workspace layout (float words) ----------
static constexpr size_t OFF_H1RAW = 0;                          // N*128
static constexpr size_t OFF_DEN1  = OFF_H1RAW + 6400000;        // N*4
static constexpr size_t OFF_M1    = OFF_DEN1  + 200000;         // N*4 (u32)
static constexpr size_t OFF_H2RAW = OFF_M1    + 200000;         // N*32
static constexpr size_t OFF_DEN2  = OFF_H2RAW + 1600000;        // N
static constexpr size_t OFF_M2    = OFF_DEN2  + 50000;          // N (u32)
static constexpr size_t OFF_BNS   = OFF_M2    + 50000;          // 128
static constexpr size_t OFF_BNQ   = OFF_BNS   + 128;            // 128
static constexpr size_t ZWORDS    = OFF_BNQ   + 128;            // zero-init region
static constexpr size_t OFF_XL1   = ZWORDS;                     // N*128
static constexpr size_t OFF_XR1   = OFF_XL1 + 6400000;          // N*128
static constexpr size_t OFF_E1    = OFF_XR1 + 6400000;          // EP*4
static constexpr size_t OFF_XL2   = OFF_E1  + 3400000;          // N*32
static constexpr size_t OFF_XR2   = OFF_XL2 + 1600000;          // N*32
static constexpr size_t OFF_E2    = OFF_XR2 + 1600000;          // EP
static constexpr size_t OFF_SS    = OFF_E2  + 850000;           // 256

extern "C" void kernel_launch(void* const* d_in, const int* in_sizes, int n_in,
                              void* d_out, int out_size, void* d_ws, size_t ws_size,
                              hipStream_t stream) {
  const float* x     = (const float*)d_in[0];
  const int*   ei    = (const int*)  d_in[1];
  const float* Wl1   = (const float*)d_in[2];
  const float* bl1   = (const float*)d_in[3];
  const float* Wr1   = (const float*)d_in[4];
  const float* br1   = (const float*)d_in[5];
  const float* att1  = (const float*)d_in[6];
  // d_in[7] = bias1 (cancels under BatchNorm)
  const float* gamma = (const float*)d_in[8];
  const float* beta  = (const float*)d_in[9];
  const float* Wl2   = (const float*)d_in[10];
  const float* bl2   = (const float*)d_in[11];
  const float* Wr2   = (const float*)d_in[12];
  const float* br2   = (const float*)d_in[13];
  const float* att2  = (const float*)d_in[14];
  const float* bias2 = (const float*)d_in[15];
  const float* Wc    = (const float*)d_in[16];
  const float* bc    = (const float*)d_in[17];
  float* out = (float*)d_out;

  float*    ws    = (float*)d_ws;
  float*    h1raw = ws + OFF_H1RAW;
  float*    den1  = ws + OFF_DEN1;
  unsigned* m1    = (unsigned*)(ws + OFF_M1);
  float*    h2raw = ws + OFF_H2RAW;
  float*    den2  = ws + OFF_DEN2;
  unsigned* m2    = (unsigned*)(ws + OFF_M2);
  float*    bns   = ws + OFF_BNS;
  float*    bnq   = ws + OFF_BNQ;
  float*    xl1   = ws + OFF_XL1;
  float*    xr1   = ws + OFF_XR1;
  float*    e1    = ws + OFF_E1;
  float*    xl2   = ws + OFF_XL2;
  float*    xr2   = ws + OFF_XR2;
  float*    e2    = ws + OFF_E2;
  float*    ss    = ws + OFF_SS;

  const int mChunks = (MTILES + MTPW - 1) / MTPW;   // 391

  // 0) zero accumulators / stats / softmax state
  zero_f32<<<2048, 256, 0, stream>>>(ws, (long)ZWORDS);

  // 1) conv1 node transforms: xl1 = x@Wl1+bl1, xr1 = x@Wr1+br1 (WMMA bf16)
  gemm_dual<<<dim3(mChunks, 16), 32, 0, stream>>>(x,
      Wl1, bl1, xl1, Wr1, br1, xr1, H1CH);

  // 2) conv1 edge softmax + aggregation
  edge1_attn<<<(EP * 4 + 255) / 256, 256, 0, stream>>>(ei, xl1, xr1, att1, e1, m1);
  edge1_exp <<<(EP * 4 + 255) / 256, 256, 0, stream>>>(ei, e1, m1, den1);
  edge1_agg <<<(EP * 32 + 255) / 256, 256, 0, stream>>>(ei, e1, den1, xl1, h1raw);

  // 3) batchnorm + ELU (in place on h1raw)
  bn_stats<<<(NNODES + 127) / 128, 128, 0, stream>>>(h1raw, bns, bnq);
  bn_final<<<1, 128, 0, stream>>>(bns, bnq, gamma, beta, ss);
  bn_apply_elu<<<(NNODES * H1CH + 255) / 256, 256, 0, stream>>>(h1raw, ss);

  // 4) conv2 node transforms (WMMA bf16)
  gemm_dual<<<dim3(mChunks, 4), 32, 0, stream>>>(h1raw,
      Wl2, bl2, xl2, Wr2, br2, xr2, HIDC);

  // 5) conv2 edge softmax + aggregation, then ELU(+bias2)
  edge2_attn<<<(EP + 255) / 256, 256, 0, stream>>>(ei, xl2, xr2, att2, e2, m2);
  edge2_exp <<<(EP + 255) / 256, 256, 0, stream>>>(ei, e2, m2, den2);
  edge2_agg <<<(EP * 8 + 255) / 256, 256, 0, stream>>>(ei, e2, den2, xl2, h2raw);
  elu2_bias <<<(NNODES * HIDC + 255) / 256, 256, 0, stream>>>(h2raw, bias2);

  // 6) classifier: out = [h2 | x] @ Wc + bc (WMMA bf16, masked 40-col store)
  gemm_cls<<<dim3(mChunks, 3), 32, 0, stream>>>(h2raw, x, Wc, bc, out);

  (void)in_sizes; (void)n_in; (void)out_size; (void)ws_size;
}
